// LeNet_75600014344539
// MI455X (gfx1250) — matrix operand order repaired and checked by reading, not compiled, not run on hardware
//
#include <hip/hip_runtime.h>

typedef __attribute__((ext_vector_type(16))) _Float16 v16h;
typedef __attribute__((ext_vector_type(8)))  _Float16 v8h;
typedef __attribute__((ext_vector_type(8)))  float    v8f;

__device__ __forceinline__ float quant_comp(float x) {
  float f = floorf(x * 128.0f);
  f = fminf(fmaxf(f, -128.0f), 127.0f);
  int qi = (int)f;
  qi |= 1;
  return (float)qi * 0.0078125f;
}

__device__ __forceinline__ v16h join16(v8h lo, v8h hi) {
  return __builtin_shufflevector(lo, hi, 0,1,2,3,4,5,6,7,8,9,10,11,12,13,14,15);
}

// ---------------------------------------------------------------------------
// K0: pack fp32 weight matrices W[Nv][Kv] (row-major) into WMMA B fragments,
// laid out so each lane's fragment is one contiguous 32B load:
//   Bpk[((nt*ktiles + kt)*32 + lane)*16 + e] = W[n][k],
//   n = nt*16 + (lane&15), k = kt*32 + (e&7) + 8*(lane>>4) + 16*(e>>3), zero-padded.
// ---------------------------------------------------------------------------
__device__ __forceinline__ void pack_one(_Float16* dst, const float* __restrict__ W,
                                         int Kv, int Nv, int ktiles, int ntiles, int i)
{
  int total = ntiles * ktiles * 512;
  if (i >= total) return;
  int e    = i & 15;
  int lane = (i >> 4) & 31;
  int kt   = (i >> 9) % ktiles;
  int nt   = i / (512 * ktiles);
  int n    = nt * 16 + (lane & 15);
  int k    = kt * 32 + (e & 7) + 8 * (lane >> 4) + 16 * (e >> 3);
  dst[i] = (k < Kv && n < Nv) ? (_Float16)W[n * Kv + k] : (_Float16)0.0f;
}

__global__ void pack_weights(const float* __restrict__ conv2_w,
                             const float* __restrict__ fc1_w,
                             const float* __restrict__ fc2_w,
                             const float* __restrict__ fc3_w,
                             _Float16* Bw2, _Float16* B1, _Float16* B2, _Float16* B3)
{
  int i = blockIdx.x * blockDim.x + threadIdx.x;
  pack_one(Bw2, conv2_w, 150, 16, 5, 1, i);   //  2560 elems
  pack_one(B1,  fc1_w,  400, 120, 13, 8, i);  // 53248 elems
  pack_one(B2,  fc2_w,  120,  84,  4, 6, i);  // 12288 elems
  pack_one(B3,  fc3_w,   84,  10,  3, 1, i);  //  1536 elems
}

// K0b: zero the 16 pad columns (400..415) of act2 rows.
__global__ void pad_act2(_Float16* __restrict__ act2, long long total)
{
  long long i = (long long)blockIdx.x * blockDim.x + threadIdx.x;
  if (i >= total) return;
  long long b = i >> 4;
  act2[b * 416 + 400 + (int)(i & 15)] = (_Float16)0.0f;
}

// ---------------------------------------------------------------------------
// K1a: quantize input -> f16 (Q1.7 values are exact in f16).
// ---------------------------------------------------------------------------
__global__ void quant_input(const float* __restrict__ x, _Float16* __restrict__ xq,
                            const int* __restrict__ qf, long long n)
{
  long long i = (long long)blockIdx.x * blockDim.x + threadIdx.x;
  if (i >= n) return;
  float v = x[i];
  if (*qf) v = quant_comp(v);
  xq[i] = (_Float16)v;
}

// ---------------------------------------------------------------------------
// K1b: conv1 (1->6, 5x5, pad 2) + bias + relu + 2x2 maxpool + quant.
// Branchless: clamped unconditional loads + select. act1: [B][6][14][14] f16.
// ---------------------------------------------------------------------------
__global__ void conv1_pool(const _Float16* __restrict__ xq,
                           const float* __restrict__ w1, const float* __restrict__ b1,
                           const int* __restrict__ qf, _Float16* __restrict__ act1,
                           long long total)
{
  long long i = (long long)blockIdx.x * blockDim.x + threadIdx.x;
  if (i >= total) return;
  int px = (int)(i % 14); long long t = i / 14;
  int py = (int)(t % 14); t /= 14;
  int c  = (int)(t % 6);  long long b = t / 6;

  const _Float16* xin = xq + b * 784;
  float rw[25];
  #pragma unroll
  for (int j = 0; j < 25; j++) rw[j] = w1[c * 25 + j];
  float bias = b1[c];

  float best = -1e30f;
  #pragma unroll
  for (int dy = 0; dy < 2; dy++)
  #pragma unroll
  for (int dx = 0; dx < 2; dx++) {
    float s = bias;
    #pragma unroll
    for (int ky = 0; ky < 5; ky++) {
      int iy = 2*py + dy + ky - 2;
      bool vy = (unsigned)iy < 28u;
      int iyc = vy ? iy : 0;
      #pragma unroll
      for (int kx = 0; kx < 5; kx++) {
        int ix = 2*px + dx + kx - 2;
        bool vv = vy && ((unsigned)ix < 28u);
        int ixc = ((unsigned)ix < 28u) ? ix : 0;
        float xv = (float)xin[iyc*28 + ixc];
        s += (vv ? xv : 0.0f) * rw[ky*5 + kx];
      }
    }
    best = fmaxf(best, fmaxf(s, 0.0f));
  }
  if (*qf) best = quant_comp(best);
  act1[b*1176 + c*196 + py*14 + px] = (_Float16)best;
}

// ---------------------------------------------------------------------------
// K2: conv2 (6->16, 5x5) + bias + relu + 2x2 maxpool + quant, WMMA via LDS.
// One 128-thread block per 16-row batch tile:
//   - stage 16x1176 act1 tile in LDS (contiguous b128 copies, read once)
//   - per pooled pixel: build four zero-padded 16x160 im2col patches in LDS
//     (batched load phase -> single dscnt wait -> store phase),
//     wave w GEMMs sub-pixel w (5 k-tiles, ds_load_b128 A fragments),
//     cross-wave max through LDS, bias+relu+quant -> act2 [B][416].
// ---------------------------------------------------------------------------
__global__ void __launch_bounds__(128)
conv2_pool_wmma(const _Float16* __restrict__ act1,
                const _Float16* __restrict__ Bpk2,
                const float* __restrict__ bias2,
                const int* __restrict__ qf,
                _Float16* __restrict__ act2)
{
  __shared__ _Float16 sAct[16 * 1176];   // 37632 B
  __shared__ _Float16 sP[4 * 16 * 160];  // 20480 B
  __shared__ float    sC[4 * 256];       //  4096 B
  __shared__ short    sTab[160];         //   320 B

  const int lane  = threadIdx.x;
  const int w     = threadIdx.y;              // wave id = sub-pixel id
  const int t     = w * 32 + lane;            // 0..127
  const int btile = blockIdx.x;
  const int half  = lane >> 4;
  const int mn    = lane & 15;

  // im2col offset table: k -> c*196 + ky*14 + kx  (k = c*25+ky*5+kx), -1 = pad
  for (int k = t; k < 160; k += 128) {
    short off = -1;
    if (k < 150) { int c = k / 25, r = k - 25 * c; off = (short)(c*196 + (r/5)*14 + (r%5)); }
    sTab[k] = off;
  }

  // stage 16 contiguous act1 rows: 37632 B = 2352 x uint4
  {
    const uint4* src = (const uint4*)(act1 + (long long)btile * 16 * 1176);
    uint4* dst = (uint4*)sAct;
    for (int j = t; j < 2352; j += 128) dst[j] = src[j];
  }

  // B fragments (shared across all 25 pixels): one 32B load each
  v16h bf[5];
  #pragma unroll
  for (int kt = 0; kt < 5; kt++)
    bf[kt] = *(const v16h*)(Bpk2 + (kt * 32 + lane) * 16);

  __syncthreads();
  const bool q = (*qf != 0);

  // This thread's 20 gather offsets (k = (t&7) + 8j) are pixel-invariant:
  // cache table entries in registers once.
  const int rowb = t >> 3, k0b = t & 7;
  short offs[20];
  #pragma unroll
  for (int j = 0; j < 20; j++) offs[j] = sTab[k0b + 8 * j];

  for (int p = 0; p < 25; p++) {
    const int py = p / 5, px = p - 5 * py;

    // build four 16x160 patches (row-major, zero-padded); batch the gather
    // loads into registers first so only one DS wait is needed per sub-pixel.
    {
      const _Float16* arow = sAct + rowb * 1176;
      _Float16* Prow0 = sP + rowb * 160 + k0b;
      #pragma unroll
      for (int s = 0; s < 4; s++) {
        const int pixoff = (2*py + (s >> 1)) * 14 + (2*px + (s & 1));
        _Float16 tmp[20];
        #pragma unroll
        for (int j = 0; j < 20; j++) {
          const short off = offs[j];
          tmp[j] = arow[pixoff + (off < 0 ? 0 : off)];
        }
        _Float16* Prow = Prow0 + s * 2560;
        #pragma unroll
        for (int j = 0; j < 20; j++) {
          Prow[8 * j] = (offs[j] < 0) ? (_Float16)0.0f : tmp[j];
        }
      }
    }
    __syncthreads();

    // GEMM: wave w -> sub-pixel w. A fragments: two ds b128 loads per k-tile.
    v8f acc = {};
    {
      const _Float16* Pw = sP + w * 2560 + mn * 160 + half * 8;
      #pragma unroll
      for (int kt = 0; kt < 5; kt++) {
        v8h lo = *(const v8h*)(Pw + kt * 32);
        v8h hi = *(const v8h*)(Pw + kt * 32 + 16);
        acc = __builtin_amdgcn_wmma_f32_16x16x32_f16(
            false, join16(lo, hi), false, bf[kt], (short)0, acc, false, false);
      }
    }
    #pragma unroll
    for (int r = 0; r < 8; r++) sC[w * 256 + (r + 8*half) * 16 + mn] = acc[r];
    __syncthreads();

    // cross-wave max-pool + bias + relu + quant + store
    #pragma unroll
    for (int u = 0; u < 2; u++) {
      const int idx = t + 128 * u;
      const int m = idx >> 4, n = idx & 15;
      float v = fmaxf(fmaxf(sC[idx], sC[256 + idx]), fmaxf(sC[512 + idx], sC[768 + idx]));
      v = fmaxf(v + bias2[n], 0.0f);
      if (q) v = quant_comp(v);
      act2[((long long)btile * 16 + m) * 416 + n * 25 + p] = (_Float16)v;
    }
    __syncthreads();
  }
}

// ---------------------------------------------------------------------------
// FC layer via WMMA. A rows are contiguous and zero-padded to 32*KT elems,
// so fragments are pure b128 loads. B pre-packed in fragment order.
// ---------------------------------------------------------------------------
template <int KT, bool RELU_QUANT, bool F16OUT>
__global__ void __launch_bounds__(128)
fc_wmma(const _Float16* __restrict__ A, int lda,
        const _Float16* __restrict__ Bpk,
        const float* __restrict__ bias, int nvalid,
        const int* __restrict__ qf,
        _Float16* __restrict__ outh, float* __restrict__ outf, int ldo)
{
  const int lane  = threadIdx.x;
  const int btile = blockIdx.x * blockDim.y + threadIdx.y;
  const int nt    = blockIdx.y;
  const int half  = lane >> 4;
  const int mn    = lane & 15;
  const int n     = nt * 16 + mn;

  const _Float16* ap = A + ((long long)btile * 16 + mn) * lda + half * 8;
  const _Float16* bp = Bpk + ((nt * KT) * 32 + lane) * 16;

  v8f acc = {};
  #pragma unroll
  for (int kt = 0; kt < KT; kt++) {
    v8h lo = *(const v8h*)(ap + kt * 32);
    v8h hi = *(const v8h*)(ap + kt * 32 + 16);
    v16h bv = *(const v16h*)(bp + kt * 512);
    acc = __builtin_amdgcn_wmma_f32_16x16x32_f16(
        false, join16(lo, hi), false, bv, (short)0, acc, false, false);
  }

  const float bs = (n < nvalid) ? bias[n] : 0.0f;
  const bool q = RELU_QUANT && (*qf != 0);
  #pragma unroll
  for (int r = 0; r < 8; r++) {
    const int M = r + 8 * half;
    const long long row = (long long)btile * 16 + M;
    float v = acc[r] + bs;
    if (RELU_QUANT) { v = fmaxf(v, 0.0f); if (q) v = quant_comp(v); }
    if (F16OUT) {
      outh[row * ldo + n] = (_Float16)v;   // padded cols hit zero B rows downstream
    } else if (n < nvalid) {
      outf[row * (long long)ldo + n] = v;
    }
  }
}

// ---------------------------------------------------------------------------
extern "C" void kernel_launch(void* const* d_in, const int* in_sizes, int n_in,
                              void* d_out, int out_size, void* d_ws, size_t ws_size,
                              hipStream_t stream)
{
  const float* x       = (const float*)d_in[0];
  const float* conv1_w = (const float*)d_in[1];
  const float* conv1_b = (const float*)d_in[2];
  const float* conv2_w = (const float*)d_in[3];
  const float* conv2_b = (const float*)d_in[4];
  const float* fc1_w   = (const float*)d_in[5];
  const float* fc1_b   = (const float*)d_in[6];
  const float* fc2_w   = (const float*)d_in[7];
  const float* fc2_b   = (const float*)d_in[8];
  const float* fc3_w   = (const float*)d_in[9];
  const float* fc3_b   = (const float*)d_in[10];
  const int*   qflag   = (const int*)d_in[11];

  const long long B = (long long)in_sizes[0] / 784;   // 16384

  // Workspace (~65 MB). Region A: xq, later reused for act2/act3/act4
  // (xq is dead after conv1_pool, strictly earlier in stream order).
  auto al = [](size_t v){ return (v + 255) & ~(size_t)255; };
  char* ws = (char*)d_ws;
  size_t sz_xq  = (size_t)B * 784 * 2;
  size_t o_act2 = 0;                                    // aliases xq @ 0
  size_t o_act3 = al(o_act2 + (size_t)B * 416 * 2);
  size_t o_act4 = al(o_act3 + (size_t)B * 128 * 2);
  size_t endA   = o_act4 + (size_t)B * 96 * 2;
  if (sz_xq > endA) endA = sz_xq;
  size_t o_act1 = al(endA);
  size_t o_Bw2  = al(o_act1 + (size_t)B * 1176 * 2);
  size_t o_B1   = al(o_Bw2 + 2560 * 2);
  size_t o_B2   = al(o_B1  + 53248 * 2);
  size_t o_B3   = al(o_B2  + 12288 * 2);
  (void)ws_size; (void)n_in; (void)out_size;

  _Float16* xq   = (_Float16*)(ws + 0);
  _Float16* act1 = (_Float16*)(ws + o_act1);
  _Float16* act2 = (_Float16*)(ws + o_act2);
  _Float16* act3 = (_Float16*)(ws + o_act3);
  _Float16* act4 = (_Float16*)(ws + o_act4);
  _Float16* Bw2  = (_Float16*)(ws + o_Bw2);
  _Float16* B1m  = (_Float16*)(ws + o_B1);
  _Float16* B2m  = (_Float16*)(ws + o_B2);
  _Float16* B3m  = (_Float16*)(ws + o_B3);

  // K0: weight repack into fragment order
  pack_weights<<<dim3((53248 + 255)/256), dim3(256), 0, stream>>>(
      conv2_w, fc1_w, fc2_w, fc3_w, Bw2, B1m, B2m, B3m);

  // K1a: quantize input
  long long nx = B * 784;
  quant_input<<<dim3((unsigned)((nx + 255)/256)), dim3(256), 0, stream>>>(
      x, xq, qflag, nx);

  // K1b: conv1 + relu + pool + quant
  long long n1 = B * 6 * 14 * 14;
  conv1_pool<<<dim3((unsigned)((n1 + 255)/256)), dim3(256), 0, stream>>>(
      xq, conv1_w, conv1_b, qflag, act1, n1);

  // zero act2 pad columns (must be after conv1: act2 aliases xq)
  long long npad = B * 16;
  pad_act2<<<dim3((unsigned)((npad + 255)/256)), dim3(256), 0, stream>>>(act2, npad);

  // K2: conv2 (WMMA, LDS-staged) + relu + pool + quant -> act2 [B,416]
  conv2_pool_wmma<<<dim3((unsigned)(B/16)), dim3(32, 4), 0, stream>>>(
      act1, Bw2, conv2_b, qflag, act2);

  dim3 blk(32, 4);
  unsigned bt4 = (unsigned)(B / 16 / 4);

  // fc1 400(416)->120 (+relu+quant) -> act3 [B,128]
  fc_wmma<13, true,  true ><<<dim3(bt4, 8), blk, 0, stream>>>(
      act2, 416, B1m, fc1_b, 120, qflag, act3, nullptr, 128);
  // fc2 120(128)->84 (+relu+quant) -> act4 [B,96]
  fc_wmma<4,  true,  true ><<<dim3(bt4, 6), blk, 0, stream>>>(
      act3, 128, B2m, fc2_b, 84, qflag, act4, nullptr, 96);
  // fc3 84(96)->10 -> d_out [B,10] fp32
  fc_wmma<3,  false, false><<<dim3(bt4, 1), blk, 0, stream>>>(
      act4, 96, B3m, fc3_b, 10, qflag, nullptr, (float*)d_out, 10);
}
